// MultiHeadAttention_48301202211428
// MI455X (gfx1250) — matrix operand order repaired and checked by reading, not compiled, not run on hardware
//
#include <hip/hip_runtime.h>

// ---------------------------------------------------------------------------
// Types for CDNA5 WMMA (wave32, 16x16x32 bf16 -> f32) and TDM descriptors
// ---------------------------------------------------------------------------
typedef __bf16 bf16_t;
typedef __attribute__((ext_vector_type(16))) __bf16 v16bf;
typedef __attribute__((ext_vector_type(8)))  float  v8f;
typedef __attribute__((ext_vector_type(4)))  unsigned int u32x4;
typedef __attribute__((ext_vector_type(8)))  int i32x8;
typedef __attribute__((ext_vector_type(4)))  int i32x4;

#define WAVE 32

__device__ __forceinline__ void wmma_bf16_acc(v16bf a, v16bf b, v8f& c) {
  // D = A(16x32) * B(32x16) + C, f32 accumulate
  c = __builtin_amdgcn_wmma_f32_16x16x32_bf16(false, a, false, b, (short)0, c,
                                              false, false);
}

// ---------------------------------------------------------------------------
// Tensor Data Mover: DMA one 32-row x 128-byte tile (row stride 2048B) from
// global memory into LDS, padding +16B per 128B row (LDS row stride 144B = 72
// bf16).  D# group layout per CDNA5 ISA.
// ---------------------------------------------------------------------------
__device__ __forceinline__ void tdm_load_tile_to_lds(const void* gsrc,
                                                     unsigned lds_off) {
  const unsigned long long ga = (unsigned long long)(size_t)gsrc;
  u32x4 g0;
  g0[0] = 1u;                                    // count=1, user descriptor
  g0[1] = lds_off;                               // LDS byte address
  g0[2] = (unsigned)ga;                          // global_addr[31:0]
  g0[3] = ((unsigned)(ga >> 32) & 0x01FFFFFFu)   // global_addr[56:32]
          | (2u << 30);                          // type = 2 ("image")
  i32x8 g1;
  g1[0] = (int)((3u << 16)     // data_size = 3 -> 8-byte units
              | (1u << 20)     // pad_enable
              | (4u << 22)     // pad_interval: 32 DWORDs (128B) between pads
              | (3u << 25));   // pad_amount:   4 DWORDs (16B)
  g1[1] = (int)(16u << 16);    // tensor_dim0 = 16 units (128B line)
  g1[2] = (int)(32u << 16);    // tensor_dim1 = 32 lines
  g1[3] = (int)(16u << 16);    // tile_dim0 = 16 units
  g1[4] = 32;                  // tile_dim1 = 32, tile_dim2 = 0
  g1[5] = 256;                 // tensor_dim0_stride = 2048B / 8B
  g1[6] = 0;
  g1[7] = 0;
  const i32x4 z4 = {0, 0, 0, 0};
#if __clang_major__ >= 23
  const i32x8 z8 = {0, 0, 0, 0, 0, 0, 0, 0};
  __builtin_amdgcn_tensor_load_to_lds(g0, g1, z4, z4, z8, 0);
#else
  __builtin_amdgcn_tensor_load_to_lds(g0, g1, z4, z4, 0);
#endif
}

// A fragment: 16x32 bf16 from row-major src (leading dim = ld elements).
__device__ __forceinline__ v16bf load_a_frag(const bf16_t* src, int ld) {
  const int lane = threadIdx.x & (WAVE - 1);
  const int row  = lane & 15;
  const int kh   = (lane >> 4) << 4;
  const bf16_t* p = src + row * ld;
  v16bf a;
#pragma unroll
  for (int v = 0; v < 4; ++v) {
    a[2 * v]     = p[kh + 2 * v];
    a[2 * v + 1] = p[kh + 2 * v + 1];
    a[8 + 2 * v]     = p[kh + 8 + 2 * v];
    a[8 + 2 * v + 1] = p[kh + 8 + 2 * v + 1];
  }
  return a;
}

// A fragment with per-K f32 bias added (used for Q + content_bias)
__device__ __forceinline__ v16bf load_a_frag_biased(const bf16_t* src, int ld,
                                                    const float* bias) {
  const int lane = threadIdx.x & (WAVE - 1);
  const int row  = lane & 15;
  const int kh   = (lane >> 4) << 4;
  const bf16_t* p = src + row * ld;
  v16bf a;
#pragma unroll
  for (int v = 0; v < 4; ++v) {
    const int k0 = kh + 2 * v;
    const int k1 = kh + 8 + 2 * v;
    a[2 * v]     = (bf16_t)((float)p[k0]     + bias[k0]);
    a[2 * v + 1] = (bf16_t)((float)p[k0 + 1] + bias[k0 + 1]);
    a[8 + 2 * v]     = (bf16_t)((float)p[k1]     + bias[k1]);
    a[8 + 2 * v + 1] = (bf16_t)((float)p[k1 + 1] + bias[k1 + 1]);
  }
  return a;
}

// B fragment: 32x16 bf16, K-major source: element (k,n) = src[k*ld + n].
__device__ __forceinline__ v16bf load_b_frag(const bf16_t* src, int ld) {
  const int lane = threadIdx.x & (WAVE - 1);
  const int n  = lane & 15;
  const int kb = (lane >> 4) << 4;
  v16bf b;
#pragma unroll
  for (int v = 0; v < 8; ++v) {
    b[2 * v]     = src[(kb + 2 * v) * ld + n];
    b[2 * v + 1] = src[(kb + 2 * v + 1) * ld + n];
  }
  return b;
}

// B fragment from N-major source (B^T row-major): element (k,n) = src[n*ld+k].
__device__ __forceinline__ v16bf load_bT_frag(const bf16_t* src, int ld) {
  const int lane = threadIdx.x & (WAVE - 1);
  const int n  = lane & 15;
  const int kb = (lane >> 4) << 4;
  const bf16_t* p = src + n * ld + kb;
  v16bf b;
#pragma unroll
  for (int v = 0; v < 16; ++v) b[v] = p[v];
  return b;
}

// ---------------------------------------------------------------------------
// Generic projection GEMM: Y(MxN) = A(MxK) @ W(KxN) + bias, bf16 WMMA compute.
// Block tile 128x64, 256 threads (8 waves), each wave: 16 rows x 64 cols.
// (A/W are f32 and converted during staging, so VALU staging, not TDM.)
// ---------------------------------------------------------------------------
template <bool A_IS_F32, bool OUT_IS_F32>
__global__ __launch_bounds__(256) void gemm_bias_kernel(
    const void* __restrict__ Ap, const float* __restrict__ W,
    const float* __restrict__ bias, void* __restrict__ Yp, int M, int N, int K) {
  constexpr int LDA = 40;
  constexpr int LDB = 72;
  __shared__ bf16_t ldsA[128 * LDA];
  __shared__ bf16_t ldsB[32 * LDB];

  const int tid  = threadIdx.x;
  const int lane = tid & (WAVE - 1);
  const int w    = tid >> 5;
  const int row0 = blockIdx.y * 128;
  const int col0 = blockIdx.x * 64;

  const v8f vzero = {0.f, 0.f, 0.f, 0.f, 0.f, 0.f, 0.f, 0.f};
  v8f acc[4];
#pragma unroll
  for (int n = 0; n < 4; ++n) acc[n] = vzero;

  for (int k0 = 0; k0 < K; k0 += 32) {
    __syncthreads();
#pragma unroll
    for (int i = 0; i < 16; ++i) {
      const int idx = tid + i * 256;
      const int r = idx >> 5, c = idx & 31;
      float v;
      if (A_IS_F32)
        v = ((const float*)Ap)[(size_t)(row0 + r) * K + k0 + c];
      else
        v = (float)((const bf16_t*)Ap)[(size_t)(row0 + r) * K + k0 + c];
      ldsA[r * LDA + c] = (bf16_t)v;
    }
#pragma unroll
    for (int i = 0; i < 8; ++i) {
      const int idx = tid + i * 256;
      const int r = idx >> 6, c = idx & 63;
      ldsB[r * LDB + c] = (bf16_t)W[(size_t)(k0 + r) * N + col0 + c];
    }
    __syncthreads();

    if (k0 + 32 < K && tid < 32)
      __builtin_prefetch(&W[(size_t)(k0 + 32 + tid) * N + col0], 0, 0);

    const v16bf a = load_a_frag(&ldsA[(16 * w) * LDA], LDA);
#pragma unroll
    for (int n = 0; n < 4; ++n) {
      const v16bf bf = load_b_frag(&ldsB[16 * n], LDB);
      wmma_bf16_acc(a, bf, acc[n]);
    }
  }

  const int rbase = (lane >> 4) << 3;
  const int col   = lane & 15;
#pragma unroll
  for (int n = 0; n < 4; ++n) {
#pragma unroll
    for (int r = 0; r < 8; ++r) {
      const int gr = row0 + 16 * w + rbase + r;
      const int gc = col0 + 16 * n + col;
      const float y = acc[n][r] + bias[gc];
      if (OUT_IS_F32)
        ((float*)Yp)[(size_t)gr * N + gc] = y;
      else
        ((bf16_t*)Yp)[(size_t)gr * N + gc] = (bf16_t)y;
    }
  }
}

// ---------------------------------------------------------------------------
// pk = pos_emb(7x1024) @ Wk(1024x1024) + bk  -> f32 (7,16,64). Tiny; scalar.
// ---------------------------------------------------------------------------
__global__ __launch_bounds__(256) void pk_kernel(const float* __restrict__ pos_emb,
                                                 const float* __restrict__ Wk,
                                                 const float* __restrict__ bk,
                                                 float* __restrict__ pk) {
  const int idx = blockIdx.x * 256 + threadIdx.x;
  if (idx >= 7 * 1024) return;
  const int r = idx >> 10, n = idx & 1023;
  float s = bk[n];
  const float* pe = pos_emb + r * 1024;
#pragma unroll 8
  for (int k = 0; k < 1024; ++k) s += pe[k] * Wk[(size_t)k * 1024 + n];
  pk[idx] = s;
}

// ---------------------------------------------------------------------------
// Fused flash attention with relative-position bias.
// Grid: (T/128, B*H). 256 threads = 8 waves; wave w owns 16 query rows
// (wave-private online softmax).  K/V tiles (32 keys) DMA'd by the Tensor
// Data Mover into double-buffered LDS; wave 0 issues next tile's TDM while
// all waves compute WMMAs on the current tile (TENSORcnt in-order per wave).
// ---------------------------------------------------------------------------
#define TQ 128
#define TK 32
#define HD 64
#define HDTOT 1024  // H*DH

__global__ __launch_bounds__(256) void flash_attn_rel_kernel(
    const bf16_t* __restrict__ qb, const bf16_t* __restrict__ kb,
    const bf16_t* __restrict__ vb, const float* __restrict__ pk,
    const float* __restrict__ content_bias, const float* __restrict__ rel_bias,
    bf16_t* __restrict__ ob, int T) {
  __shared__ float  smallLds[TQ][8];       // (q row in tile) x 7 rel logits
  __shared__ bf16_t ldsK[2][TK][72];       // double-buffered, TDM-padded rows
  __shared__ bf16_t ldsV[2][TK][72];
  __shared__ bf16_t ldsP[8][16 * 40];      // wave-private P staging (16x32+pad)

  const int h  = blockIdx.y & 15;
  const int b  = blockIdx.y >> 4;
  const int q0 = blockIdx.x * TQ;
  const int tid  = threadIdx.x;
  const int lane = tid & (WAVE - 1);
  const int w    = tid >> 5;
  const int rbase = (lane >> 4) << 3;
  const int col   = lane & 15;

  const bf16_t* qhb = qb + ((size_t)b * T + q0) * HDTOT + h * HD;
  const bf16_t* Kh  = kb + ((size_t)b * T) * HDTOT + h * HD;
  const bf16_t* Vh  = vb + ((size_t)b * T) * HDTOT + h * HD;

  // Prologue: wave 0 kicks the TDM for tile 0 into buffer 0.
  if (w == 0) {
    tdm_load_tile_to_lds(Kh, (unsigned)(size_t)&ldsK[0][0][0]);
    tdm_load_tile_to_lds(Vh, (unsigned)(size_t)&ldsV[0][0][0]);
  }

  // ---- relative term: small[t][r] = sum_d (q[t][d] + rel_bias[h][d]) * pk[r][h][d]
  const float* rb = rel_bias + h * HD;
  for (int idx = tid; idx < TQ * 7; idx += 256) {
    const int t = idx / 7, r = idx % 7;
    const bf16_t* qrow = qhb + (size_t)t * HDTOT;
    const float* pkr = pk + (r * 16 + h) * HD;
    float s = 0.f;
#pragma unroll 8
    for (int d = 0; d < HD; ++d) s += ((float)qrow[d] + rb[d]) * pkr[d];
    smallLds[t][r] = s;
  }

  // ---- loop-invariant Q(+content_bias) A fragments (d 0..31, 32..63)
  const float* u = content_bias + h * HD;
  const v16bf qa0 = load_a_frag_biased(qhb + (size_t)(16 * w) * HDTOT, HDTOT, u);
  const v16bf qa1 = load_a_frag_biased(qhb + (size_t)(16 * w) * HDTOT + 32, HDTOT, u + 32);

  const v8f vzero = {0.f, 0.f, 0.f, 0.f, 0.f, 0.f, 0.f, 0.f};
  v8f o[4];
#pragma unroll
  for (int n = 0; n < 4; ++n) o[n] = vzero;
  v8f m_run, l_run;
#pragma unroll
  for (int r = 0; r < 8; ++r) { m_run[r] = -__builtin_inff(); l_run[r] = 0.f; }

  const int trow0 = q0 + 16 * w + rbase;

  for (int kt = 0; kt < T; kt += TK) {
    const int cur = (kt >> 5) & 1;
    const int nxt = cur ^ 1;

    __syncthreads();  // all waves done reading buffer `nxt` (tile kt-32)
    if (w == 0) {     // wave-uniform branch: only wave 0 runs the TDM ops
      if (kt + TK < T) {
        tdm_load_tile_to_lds(Kh + (size_t)(kt + TK) * HDTOT,
                             (unsigned)(size_t)&ldsK[nxt][0][0]);
        tdm_load_tile_to_lds(Vh + (size_t)(kt + TK) * HDTOT,
                             (unsigned)(size_t)&ldsV[nxt][0][0]);
        __builtin_amdgcn_s_wait_tensorcnt(2);  // tile kt resident, next in flight
      } else {
        __builtin_amdgcn_s_wait_tensorcnt(0);
      }
    }
    __syncthreads();  // tile kt visible to all waves

    // ---- S = (Q+u) @ K^T : two 16x16 tiles, K-dim 64 = 2 WMMA steps each
    v8f s0 = vzero, s1 = vzero;
    wmma_bf16_acc(qa0, load_bT_frag(&ldsK[cur][0][0],  72), s0);
    wmma_bf16_acc(qa1, load_bT_frag(&ldsK[cur][0][32], 72), s0);
    wmma_bf16_acc(qa0, load_bT_frag(&ldsK[cur][16][0],  72), s1);
    wmma_bf16_acc(qa1, load_bT_frag(&ldsK[cur][16][32], 72), s1);

    // ---- add relative logits + scale 1/sqrt(DH)
#pragma unroll
    for (int r = 0; r < 8; ++r) {
      const int tq = trow0 + r;
      int d0 = tq - (kt + col);
      int d1 = tq - (kt + 16 + col);
      d0 = d0 < -3 ? -3 : (d0 > 3 ? 3 : d0);
      d1 = d1 < -3 ? -3 : (d1 > 3 ? 3 : d1);
      s0[r] = (s0[r] + smallLds[tq - q0][d0 + 3]) * 0.125f;
      s1[r] = (s1[r] + smallLds[tq - q0][d1 + 3]) * 0.125f;
    }

    // ---- online softmax (rows live in 16-lane halves; xor 1..8 stays in-half)
    v8f mt;
#pragma unroll
    for (int r = 0; r < 8; ++r) mt[r] = fmaxf(s0[r], s1[r]);
#pragma unroll
    for (int off = 1; off <= 8; off <<= 1)
#pragma unroll
      for (int r = 0; r < 8; ++r) mt[r] = fmaxf(mt[r], __shfl_xor(mt[r], off, WAVE));

    v8f p0, p1, rsum, alpha;
#pragma unroll
    for (int r = 0; r < 8; ++r) {
      const float mn = fmaxf(m_run[r], mt[r]);
      alpha[r] = __expf(m_run[r] - mn);
      m_run[r] = mn;
      p0[r] = __expf(s0[r] - mn);
      p1[r] = __expf(s1[r] - mn);
      rsum[r] = p0[r] + p1[r];
    }
#pragma unroll
    for (int off = 1; off <= 8; off <<= 1)
#pragma unroll
      for (int r = 0; r < 8; ++r) rsum[r] += __shfl_xor(rsum[r], off, WAVE);
#pragma unroll
    for (int r = 0; r < 8; ++r) l_run[r] = l_run[r] * alpha[r] + rsum[r];
#pragma unroll
    for (int n = 0; n < 4; ++n)
#pragma unroll
      for (int r = 0; r < 8; ++r) o[n][r] *= alpha[r];

    // ---- P: C-layout -> A-layout via wave-private LDS (in-order LDS per wave)
    bf16_t* pp = &ldsP[w][0];
#pragma unroll
    for (int r = 0; r < 8; ++r) {
      pp[(rbase + r) * 40 + col]      = (bf16_t)p0[r];
      pp[(rbase + r) * 40 + col + 16] = (bf16_t)p1[r];
    }
    const v16bf pf = load_a_frag(pp, 40);

    // ---- O += P(16x32) @ V(32x64)
#pragma unroll
    for (int n = 0; n < 4; ++n)
      wmma_bf16_acc(pf, load_b_frag(&ldsV[cur][0][16 * n], 72), o[n]);
  }

  // ---- epilogue: normalize, store bf16 attn output (B,T,H*DH)
  v8f inv;
#pragma unroll
  for (int r = 0; r < 8; ++r) inv[r] = 1.0f / l_run[r];
  bf16_t* out = ob + ((size_t)b * T + q0 + 16 * w) * HDTOT + h * HD;
#pragma unroll
  for (int n = 0; n < 4; ++n)
#pragma unroll
    for (int r = 0; r < 8; ++r)
      out[(size_t)(rbase + r) * HDTOT + 16 * n + col] = (bf16_t)(o[n][r] * inv[r]);
}

// ---------------------------------------------------------------------------
// Host launcher
// ---------------------------------------------------------------------------
extern "C" void kernel_launch(void* const* d_in, const int* in_sizes, int n_in,
                              void* d_out, int out_size, void* d_ws, size_t ws_size,
                              hipStream_t stream) {
  (void)in_sizes; (void)n_in; (void)out_size; (void)ws_size;
  const int B = 2, T = 2048, D = 1024, NHD = 1024;  // H*DH
  const int M = B * T;

  const float* query = (const float*)d_in[0];
  const float* key   = (const float*)d_in[1];
  const float* value = (const float*)d_in[2];
  const float* Wq = (const float*)d_in[3];
  const float* bq = (const float*)d_in[4];
  const float* Wk = (const float*)d_in[5];
  const float* bk = (const float*)d_in[6];
  const float* Wv = (const float*)d_in[7];
  const float* bv = (const float*)d_in[8];
  const float* Wo = (const float*)d_in[9];
  const float* bo = (const float*)d_in[10];
  const float* content_bias  = (const float*)d_in[11];
  const float* relative_bias = (const float*)d_in[12];
  const float* pos_emb       = (const float*)d_in[13];

  const size_t TOK = (size_t)M * NHD;
  bf16_t* q_bf = (bf16_t*)d_ws;
  bf16_t* k_bf = q_bf + TOK;
  bf16_t* v_bf = k_bf + TOK;
  bf16_t* a_bf = v_bf + TOK;
  float*  pkbf = (float*)(a_bf + TOK);  // 7*1024 f32

  const dim3 blk(256);
  const dim3 gproj(NHD / 64, M / 128);  // (16, 32)

  // 1) Q/K/V projections: f32 in, bf16 out
  gemm_bias_kernel<true, false><<<gproj, blk, 0, stream>>>(query, Wq, bq, q_bf, M, NHD, D);
  gemm_bias_kernel<true, false><<<gproj, blk, 0, stream>>>(key,   Wk, bk, k_bf, M, NHD, D);
  gemm_bias_kernel<true, false><<<gproj, blk, 0, stream>>>(value, Wv, bv, v_bf, M, NHD, D);

  // 2) pk = pos_emb @ Wk + bk
  pk_kernel<<<(7 * 1024 + 255) / 256, blk, 0, stream>>>(pos_emb, Wk, bk, pkbf);

  // 3) Fused flash attention (TDM-fed, double-buffered) -> bf16 attn
  const dim3 gattn(T / TQ, B * 16);  // (16, 32)
  flash_attn_rel_kernel<<<gattn, blk, 0, stream>>>(q_bf, k_bf, v_bf, pkbf,
                                                   content_bias, relative_bias,
                                                   a_bf, T);

  // 4) Output projection: bf16 in, f32 out into d_out
  gemm_bias_kernel<false, true><<<gproj, blk, 0, stream>>>(a_bf, Wo, bo, (float*)d_out,
                                                           M, D, NHD);
}
